// ExllamaV3TorchLinear_40767829574208
// MI455X (gfx1250) — matrix operand order repaired and checked by reading, not compiled, not run on hardware
//
#include <hip/hip_runtime.h>

typedef __attribute__((ext_vector_type(16))) _Float16 v16h;
typedef __attribute__((ext_vector_type(8)))  float    v8f;

#define IN_F  4096
#define OUT_F 4096
#define KSTEP 64
#define ROWH  80                       // 64 data halves + 16 pad halves = 160 B row stride
#define BUFB  (128 * ROWH * 2)         // bytes per LDS buffer per matrix (20480)

// ---------------------------------------------------------------------------
// Kernel 1: x' = FHT128(x * suh)  -> fp16   (left Hadamard + input signs)
// ---------------------------------------------------------------------------
__global__ __launch_bounds__(128) void had_x_kernel(const float* __restrict__ x,
                                                    const float* __restrict__ suh,
                                                    _Float16* __restrict__ xh) {
  __shared__ float buf[128];
  const int t   = threadIdx.x;
  const int row = blockIdx.x;   // 0..4095 tokens
  const int blk = blockIdx.y;   // 0..31   128-wide K blocks
  const size_t base = (size_t)row * IN_F + (size_t)blk * 128;

  float v = x[base + t] * suh[blk * 128 + t];
  buf[t] = v;
  __syncthreads();
  #pragma unroll
  for (int s = 1; s < 128; s <<= 1) {
    const float a = buf[t ^ s];
    const float b = buf[t];
    __syncthreads();
    v = (t & s) ? (a - b) : (b + a);
    buf[t] = v;
    __syncthreads();
  }
  xh[base + t] = (_Float16)(v * 0.08838834764831845f);  // 1/sqrt(128)
}

// ---------------------------------------------------------------------------
// Kernel 2: trellis dequant -> W_inner^T (fp16, [OUT_F][IN_F])
// One wave32 per 16x16 tile; lane t produces the fragment-permutation's 8
// elements (mirrors the reference perm exactly).
// ---------------------------------------------------------------------------
__device__ __forceinline__ float exl3_lut(unsigned enc) {
  unsigned raw = enc * 89226354u + 64248484u;          // wraps mod 2^32
  raw = 996162400u ^ (raw & 2415890431u);
  unsigned short lb = (unsigned short)(raw & 0xFFFFu);
  unsigned short hb = (unsigned short)(raw >> 16);
  _Float16 lo, hi;
  __builtin_memcpy(&lo, &lb, 2);
  __builtin_memcpy(&hi, &hb, 2);
  return (float)lo + (float)hi;
}

__global__ __launch_bounds__(256) void dequant_kernel(const int* __restrict__ trellis,
                                                      _Float16* __restrict__ wt) {
  __shared__ int words[8][64];
  const int tid  = threadIdx.x;
  const int wave = tid >> 5;
  const int lane = tid & 31;
  const int tile = blockIdx.x * 8 + wave;   // 0..65535
  const int kt = tile >> 8;
  const int nt = tile & 255;

  const int* wp = trellis + (size_t)tile * 64;
  words[wave][2 * lane]     = wp[2 * lane]     & 0xFFFF;
  words[wave][2 * lane + 1] = wp[2 * lane + 1] & 0xFFFF;
  __syncthreads();

  const int i0 = lane * 8;
  int sym[11];
  #pragma unroll
  for (int j = 0; j < 11; ++j) {
    const int ii = (i0 + j - 3) & 255;
    const int w  = words[wave][(ii >> 2) ^ 1];
    sym[j] = (w >> (12 - 4 * (ii & 3))) & 15;
  }

  const int r0 = (lane & 3) * 2;
  const int c0 = lane >> 2;
  const int rows[4] = { r0, r0 + 1, r0 + 8, r0 + 9 };

  #pragma unroll
  for (int j = 0; j < 8; ++j) {
    const unsigned enc = (unsigned)sym[j + 3]
                       | ((unsigned)sym[j + 2] << 4)
                       | ((unsigned)sym[j + 1] << 8)
                       | ((unsigned)sym[j]     << 12);
    const float v = exl3_lut(enc);
    const int row = rows[j & 3];
    const int col = c0 + ((j < 4) ? 0 : 8);
    wt[(size_t)(nt * 16 + col) * IN_F + (size_t)(kt * 16 + row)] = (_Float16)v;
  }
}

// ---------------------------------------------------------------------------
// Kernel 3: t = x' @ W_inner  via v_wmma_f32_16x16x32_f16.
// 128x128 tile / 8-wave WG; per wave 2x4 subtiles. K-step 64, double-buffered
// LDS filled with global_load_async_to_lds_b128 (ASYNCcnt-tracked).
// ---------------------------------------------------------------------------
__global__ __launch_bounds__(256) void gemm_kernel(const _Float16* __restrict__ A,
                                                   const _Float16* __restrict__ Bt,
                                                   float* __restrict__ C) {
  constexpr int K = IN_F;
  constexpr int N = OUT_F;
  __shared__ __align__(128) _Float16 sA[2][128][ROWH];
  __shared__ __align__(128) _Float16 sB[2][128][ROWH];

  const int tid  = threadIdx.x;
  const int wave = tid >> 5;
  const int lane = tid & 31;
  const int bm = blockIdx.y * 128;
  const int bn = blockIdx.x * 128;
  const int wr = wave >> 1;      // 0..3
  const int wc = wave & 1;       // 0..1
  const int half = lane >> 4;    // 0/1
  const int l16  = lane & 15;

  // staging: one thread per half-row (64 B) of the 128x64h slab
  const int lrow = tid >> 1;     // 0..127
  const int lseg = tid & 1;      // 0..1
  const char* gA = (const char*)(A  + (size_t)(bm + lrow) * K) + lseg * 64;
  const char* gB = (const char*)(Bt + (size_t)(bn + lrow) * K) + lseg * 64;
  // LDS byte offset = low 32 bits of the generic pointer (LDS aperture)
  const unsigned ldsA0 = (unsigned)(unsigned long long)&sA[0][0][0]
                       + (unsigned)(lrow * (ROWH * 2) + lseg * 64);
  const unsigned ldsB0 = (unsigned)(unsigned long long)&sB[0][0][0]
                       + (unsigned)(lrow * (ROWH * 2) + lseg * 64);

  auto issue = [&](int k0, int buf) {
    const unsigned long long a = (unsigned long long)(gA + (size_t)k0 * 2);
    const unsigned long long b = (unsigned long long)(gB + (size_t)k0 * 2);
    const unsigned la = ldsA0 + (unsigned)(buf * BUFB);
    const unsigned lb = ldsB0 + (unsigned)(buf * BUFB);
    #pragma unroll
    for (int j = 0; j < 4; ++j) {
      asm volatile("global_load_async_to_lds_b128 %0, %1, off"
                   :: "v"(la + j * 16), "v"(a + j * 16) : "memory");
      asm volatile("global_load_async_to_lds_b128 %0, %1, off"
                   :: "v"(lb + j * 16), "v"(b + j * 16) : "memory");
    }
  };

  constexpr int NS = K / KSTEP;  // 64 slabs
  issue(0, 0);
  issue(KSTEP, 1);

  v8f acc[2][4] = {};

  for (int i = 0; i < NS; ++i) {
    if (i + 1 < NS) asm volatile("s_wait_asynccnt 8" ::: "memory");
    else            asm volatile("s_wait_asynccnt 0" ::: "memory");
    __syncthreads();

    const int buf = i & 1;
    #pragma unroll
    for (int kc = 0; kc < 2; ++kc) {
      // A fragments: lane half splits K contiguously 0-15 / 16-31 (matches B)
      v16h afrag[2];
      #pragma unroll
      for (int mi = 0; mi < 2; ++mi) {
        const unsigned* rowp =
            (const unsigned*)&sA[buf][wr * 32 + mi * 16 + l16][kc * 32];
        unsigned* dst = (unsigned*)&afrag[mi];
        #pragma unroll
        for (int p = 0; p < 8; ++p)
          dst[p] = rowp[p + half * 8];
      }
      // B fragments: lane half splits K contiguously 0-15 / 16-31
      v16h bfrag[4];
      #pragma unroll
      for (int ni = 0; ni < 4; ++ni)
        bfrag[ni] = *(const v16h*)&sB[buf][wc * 64 + ni * 16 + l16][kc * 32 + half * 16];

      #pragma unroll
      for (int mi = 0; mi < 2; ++mi)
        #pragma unroll
        for (int ni = 0; ni < 4; ++ni)
          acc[mi][ni] = __builtin_amdgcn_wmma_f32_16x16x32_f16(
              false, afrag[mi], false, bfrag[ni],
              (short)0, acc[mi][ni], false, false);
    }
    __syncthreads();
    if (i + 2 < NS) issue((i + 2) * KSTEP, buf);  // buf (i+2)&1 == i&1
  }

  // C fragment layout: VGPR v, lanes 0-15 -> (M=v, N=l16); lanes 16-31 -> M=v+8
  #pragma unroll
  for (int mi = 0; mi < 2; ++mi)
    #pragma unroll
    for (int ni = 0; ni < 4; ++ni) {
      const int col = bn + wc * 64 + ni * 16 + l16;
      #pragma unroll
      for (int v = 0; v < 8; ++v) {
        const int row = bm + wr * 32 + mi * 16 + v + half * 8;
        C[(size_t)row * N + col] = acc[mi][ni][v];
      }
    }
}

// ---------------------------------------------------------------------------
// Kernel 4: y = FHT128(t) * svh  (right Hadamard + output signs), in place.
// ---------------------------------------------------------------------------
__global__ __launch_bounds__(128) void had_out_kernel(float* __restrict__ y,
                                                      const float* __restrict__ svh) {
  __shared__ float buf[128];
  const int t = threadIdx.x;
  const size_t base = (size_t)blockIdx.x * OUT_F + (size_t)blockIdx.y * 128;

  float v = y[base + t];
  buf[t] = v;
  __syncthreads();
  #pragma unroll
  for (int s = 1; s < 128; s <<= 1) {
    const float a = buf[t ^ s];
    const float b = buf[t];
    __syncthreads();
    v = (t & s) ? (a - b) : (b + a);
    buf[t] = v;
    __syncthreads();
  }
  y[base + t] = v * 0.08838834764831845f * svh[blockIdx.y * 128 + t];
}

// ---------------------------------------------------------------------------
extern "C" void kernel_launch(void* const* d_in, const int* in_sizes, int n_in,
                              void* d_out, int out_size, void* d_ws, size_t ws_size,
                              hipStream_t stream) {
  (void)in_sizes; (void)n_in; (void)out_size; (void)ws_size;
  const float* x       = (const float*)d_in[0];
  const int*   trellis = (const int*)d_in[1];
  const float* suh     = (const float*)d_in[2];
  const float* svh     = (const float*)d_in[3];
  float* y = (float*)d_out;

  _Float16* xh = (_Float16*)d_ws;                                         // 32 MB
  _Float16* wt = (_Float16*)((char*)d_ws + (size_t)IN_F * OUT_F * 2);     // 32 MB

  had_x_kernel  <<<dim3(4096, 32), 128, 0, stream>>>(x, suh, xh);
  dequant_kernel<<<dim3(8192),     256, 0, stream>>>(trellis, wt);
  gemm_kernel   <<<dim3(32, 32),   256, 0, stream>>>(xh, wt, y);
  had_out_kernel<<<dim3(4096, 32), 128, 0, stream>>>(y, svh);
}
